// Attention_31533649887340
// MI455X (gfx1250) — matrix operand order repaired and checked by reading, not compiled, not run on hardware
//
#include <hip/hip_runtime.h>
#include <hip/hip_bf16.h>
#include <math.h>
#include <stdint.h>

typedef __attribute__((ext_vector_type(16))) __bf16 v16bf;
typedef __attribute__((ext_vector_type(8)))  float  v8f;
typedef __attribute__((ext_vector_type(4)))  unsigned int u32x4;
typedef __attribute__((ext_vector_type(8)))  int    i32x8;
typedef __attribute__((ext_vector_type(4)))  int    i32x4;

#define B_  32
#define S_  4096
#define D_  512
#define U_  512

#define BROW_BYTES 1056            // 1024 data + 32 pad (TDM pad: 8 dwords / 1024B)
#define BBUF_BYTES (16 * BROW_BYTES)

// ---------------------------------------------------------------------------
// Kernel 1: qpb[b][u] = query[b,:] . Ws[:,u] + bs[u] + bt[u]   (folds biases)
// ---------------------------------------------------------------------------
__global__ void qproj_kernel(const float* __restrict__ query,
                             const float* __restrict__ Ws,
                             const float* __restrict__ bs,
                             const float* __restrict__ bt,
                             float* __restrict__ qpb) {
    const int b = blockIdx.x;
    const int t = threadIdx.x;
    for (int u = t; u < U_; u += 256) {
        float acc = 0.0f;
        for (int d = 0; d < D_; ++d)
            acc = fmaf(query[b * D_ + d], Ws[d * U_ + u], acc);
        qpb[b * U_ + u] = acc + bs[u] + bt[u];
    }
}

// ---------------------------------------------------------------------------
// Kernel 1b: Wtb[u][k] = bf16(Wt[k][u])  (transpose + convert, one time).
// Makes each WMMA B-tile a contiguous 16 KB block -> TDM-copyable.
// ---------------------------------------------------------------------------
__global__ void wtb_kernel(const float* __restrict__ Wt,
                           __bf16* __restrict__ Wtb) {
    __shared__ __bf16 tile[32][33];
    const int tx = threadIdx.x & 31;
    const int ty = threadIdx.x >> 5;          // 8 rows per pass
    const int k0 = blockIdx.x * 32;
    const int u0 = blockIdx.y * 32;
#pragma unroll
    for (int i = 0; i < 4; ++i) {
        const int k = ty + i * 8;
        tile[k][tx] = (__bf16)Wt[(k0 + k) * U_ + (u0 + tx)];
    }
    __syncthreads();
#pragma unroll
    for (int i = 0; i < 4; ++i) {
        const int u = ty + i * 8;
        Wtb[(size_t)(u0 + u) * D_ + (k0 + tx)] = tile[tx][u];
    }
}

// ---------------------------------------------------------------------------
// Kernel 2: fused score[b,s] = sum_u tanh(values[b,s,:].Wt[:,u] + qpb[b,u])*Wa[u]
// A tile (16 s-rows x 512 k, bf16) lives in 128 VGPRs/lane for the whole loop.
// B tiles are DMA'd from Wtb into double-buffered LDS by the Tensor Data Mover,
// overlapped with the 16-WMMA K-loop; sync = s_wait_tensorcnt + one barrier.
// ---------------------------------------------------------------------------
__device__ __forceinline__ void tdm_load_b_tile(const __bf16* Wtb, int nt,
                                                unsigned lds_off) {
    const unsigned long long ga =
        (unsigned long long)(uintptr_t)(Wtb + (size_t)(nt * 16) * D_);
    u32x4 g0 = {0u, 0u, 0u, 0u};
    i32x8 g1 = {0, 0, 0, 0, 0, 0, 0, 0};
    i32x4 g2 = {0, 0, 0, 0};
    i32x4 g3 = {0, 0, 0, 0};
    // group0: count=1 | lds_addr | global_addr[56:0] | type=2
    g0[0] = 1u;
    g0[1] = lds_off;
    g0[2] = (unsigned)ga;
    g0[3] = (unsigned)((ga >> 32) & 0x01FFFFFFu) | (2u << 30);
    // group1: data_size=8B, pad_enable, pad_interval=7 (1024B), pad_amount=7 (32B)
    const unsigned td0 = 65536u;   // whole Wtb = 512KB / 8B
    g1[0] = (int)((3u << 16) | (1u << 20) | (7u << 22) | (7u << 25));
    g1[1] = (int)((td0 & 0xFFFFu) << 16);                      // tensor_dim0 lo16
    g1[2] = (int)(((td0 >> 16) & 0xFFFFu) | (1u << 16));       // td0 hi16 | tensor_dim1=1
    g1[3] = (int)(2048u << 16);                                // tile_dim0 = 16KB/8B
    g1[4] = 1;                                                 // tile_dim1 = 1 (1-D copy)
    g1[5] = (int)td0;                                          // tensor_dim0_stride lo32
#if __clang_major__ >= 23
    i32x8 g4 = {0, 0, 0, 0, 0, 0, 0, 0};
    __builtin_amdgcn_tensor_load_to_lds(g0, g1, g2, g3, g4, 0);
#else
    __builtin_amdgcn_tensor_load_to_lds(g0, g1, g2, g3, 0);
#endif
}

__global__ __launch_bounds__(256, 1)
void score_kernel(const float* __restrict__ values,
                  const __bf16* __restrict__ Wtb,
                  const float* __restrict__ Wa,
                  const float* __restrict__ ba,
                  const float* __restrict__ qpb,
                  float* __restrict__ score_out) {   // -> d_out weights region [B,S]
    __shared__ __align__(16) unsigned char BldsRaw[2][BBUF_BYTES];

    const int b    = blockIdx.x;
    const int s0   = blockIdx.y * 128;
    const int t    = threadIdx.x;
    const int lane = t & 31;
    const int wv   = t >> 5;        // wave id 0..7 -> 16-row A tile
    const int hf   = lane >> 4;     // half of wave (ISA A/B layout)
    const int mrow = lane & 15;

    // kick DMA of first B tile immediately (overlaps with A preload)
    if (t == 0)
        tdm_load_b_tile(Wtb, 0, (unsigned)(uintptr_t)&BldsRaw[0][0]);

    // ---- preload A tile (16 rows x 512 k, bf16) into 128 VGPRs per lane ----
    const float* arow = values + ((size_t)b * S_ + (size_t)(s0 + wv * 16 + mrow)) * D_;
    v16bf afrag[16];
#pragma unroll
    for (int kt = 0; kt < 16; ++kt) {
        const int k0 = kt * 32 + hf * 8;
#pragma unroll
        for (int e = 0; e < 8; ++e) {
            afrag[kt][e]     = (__bf16)arow[k0 + e];        // K = kt*32 + hf*8 + e
            afrag[kt][e + 8] = (__bf16)arow[k0 + 16 + e];   // K = kt*32 + 16 + hf*8 + e
        }
    }

    float sacc[8] = {0.f, 0.f, 0.f, 0.f, 0.f, 0.f, 0.f, 0.f};

    for (int nt = 0; nt < 32; ++nt) {            // 32 unit-tiles of 16
        const int cur = nt & 1;
        if (t == 0)
            __builtin_amdgcn_s_wait_tensorcnt(0);   // buf[cur] DMA complete
        __syncthreads();                            // visible to all waves
        if (t == 0 && nt < 31)                      // prefetch next tile into other buf
            tdm_load_b_tile(Wtb, nt + 1,
                            (unsigned)(uintptr_t)&BldsRaw[cur ^ 1][0]);

        v8f c = {};
        const unsigned char* bbase = &BldsRaw[cur][0] + mrow * BROW_BYTES + hf * 16;
#pragma unroll
        for (int kt = 0; kt < 16; ++kt) {
            const __bf16* bp = (const __bf16*)(bbase + kt * 64);
            v16bf bfrag;
#pragma unroll
            for (int e = 0; e < 16; ++e)
                bfrag[e] = bp[e < 8 ? e : e + 8];   // K = kt*32 + hf*8 + e (+8 for e>=8), N = mrow
            c = __builtin_amdgcn_wmma_f32_16x16x32_bf16(
                    false, afrag[kt], false, bfrag, (short)0, c, false, false);
        }

        // epilogue: C layout -> lane holds (M = v + hf*8, N = mrow)
        const int   u   = nt * 16 + mrow;
        const float qb  = qpb[b * U_ + u];
        const float wav = Wa[u];
#pragma unroll
        for (int v = 0; v < 8; ++v)
            sacc[v] += tanhf(c[v] + qb) * wav;
    }

    // sum over N: reduce across the 16 lanes of each half
#pragma unroll
    for (int v = 0; v < 8; ++v) {
        float x = sacc[v];
        x += __shfl_xor(x, 1, 32);
        x += __shfl_xor(x, 2, 32);
        x += __shfl_xor(x, 4, 32);
        x += __shfl_xor(x, 8, 32);
        sacc[v] = x;
    }
    if (mrow == 0) {
        const float bav = ba[0];
#pragma unroll
        for (int v = 0; v < 8; ++v) {
            const int srow = s0 + wv * 16 + hf * 8 + v;   // M = v + hf*8
            score_out[b * S_ + srow] = sacc[v] + bav;
        }
    }
}

// ---------------------------------------------------------------------------
// Kernel 3: softmax over S, in place on the weights region of d_out
// ---------------------------------------------------------------------------
__global__ void softmax_kernel(float* __restrict__ wts) {
    __shared__ float red[256];
    const int b = blockIdx.x;
    const int t = threadIdx.x;
    float* p = wts + b * S_;

    float m = -1e30f;
    for (int i = t; i < S_; i += 256) m = fmaxf(m, p[i]);
    red[t] = m; __syncthreads();
    for (int off = 128; off > 0; off >>= 1) {
        if (t < off) red[t] = fmaxf(red[t], red[t + off]);
        __syncthreads();
    }
    const float mx = red[0]; __syncthreads();

    float s = 0.0f;
    for (int i = t; i < S_; i += 256) s += __expf(p[i] - mx);
    red[t] = s; __syncthreads();
    for (int off = 128; off > 0; off >>= 1) {
        if (t < off) red[t] += red[t + off];
        __syncthreads();
    }
    const float inv = 1.0f / red[0]; __syncthreads();

    for (int i = t; i < S_; i += 256) p[i] = __expf(p[i] - mx) * inv;
}

// ---------------------------------------------------------------------------
// Kernel 4: zero the attention output region (harness poisons d_out)
// ---------------------------------------------------------------------------
__global__ void zero_att_kernel(float* __restrict__ att) {
    att[blockIdx.x * 256 + threadIdx.x] = 0.0f;
}

// ---------------------------------------------------------------------------
// Kernel 5: attention[b,d] = sum_s w[b,s] * values[b,s,d]  (S-chunked + atomics)
// ---------------------------------------------------------------------------
__global__ void attend_kernel(const float* __restrict__ values,
                              const float* __restrict__ wts,
                              float* __restrict__ att) {
    const int b  = blockIdx.x;
    const int s0 = blockIdx.y * 256;
    const int d0 = threadIdx.x * 2;

    const float* vb = values + ((size_t)b * S_ + s0) * D_ + d0;
    const float* wp = wts + b * S_ + s0;

    float ax = 0.0f, ay = 0.0f;
    for (int s = 0; s < 256; ++s) {
        const float w = wp[s];
        ax = fmaf(w, vb[0], ax);
        ay = fmaf(w, vb[1], ay);
        vb += D_;
    }
    atomicAdd(&att[b * D_ + d0],     ax);
    atomicAdd(&att[b * D_ + d0 + 1], ay);
}

// ---------------------------------------------------------------------------
extern "C" void kernel_launch(void* const* d_in, const int* in_sizes, int n_in,
                              void* d_out, int out_size, void* d_ws, size_t ws_size,
                              hipStream_t stream) {
    const float* query  = (const float*)d_in[0];
    const float* values = (const float*)d_in[1];
    const float* Ws     = (const float*)d_in[2];
    const float* bs     = (const float*)d_in[3];
    const float* Wt     = (const float*)d_in[4];
    const float* bt     = (const float*)d_in[5];
    const float* Wa     = (const float*)d_in[6];
    const float* ba     = (const float*)d_in[7];

    float* out = (float*)d_out;
    float* att = out;               // [B, D]    = 16384 floats
    float* wts = out + B_ * D_;     // [B, S, 1] = 131072 floats

    float*  qpb = (float*)d_ws;                            // 64 KB
    __bf16* Wtb = (__bf16*)((char*)d_ws + 64 * 1024);      // 512 KB, [U][D] bf16

    qproj_kernel  <<<dim3(B_),           dim3(256), 0, stream>>>(query, Ws, bs, bt, qpb);
    wtb_kernel    <<<dim3(16, 16),       dim3(256), 0, stream>>>(Wt, Wtb);
    score_kernel  <<<dim3(B_, S_ / 128), dim3(256), 0, stream>>>(values, Wtb, Wa, ba, qpb, wts);
    softmax_kernel<<<dim3(B_),           dim3(256), 0, stream>>>(wts);
    zero_att_kernel<<<dim3((B_ * D_) / 256), dim3(256), 0, stream>>>(att);
    attend_kernel <<<dim3(B_, S_ / 256), dim3(256), 0, stream>>>(values, wts, att);
}
